// SGC_75204877353222
// MI455X (gfx1250) — matrix operand order repaired and checked by reading, not compiled, not run on hardware
//
#include <hip/hip_runtime.h>
#include <hip/hip_bf16.h>

// ---------------------------------------------------------------------------
// SGC (Simplified Graph Convolution) on MI455X / gfx1250.
//  - 2 sparse propagation hops via f32 global atomics (h buffers fit in 192MB L2)
//  - final dense projection emb = h @ W^T + b with v_wmma_f32_16x16x32_f16
//  - row-wise log_softmax with wave32 shuffle reductions
// ---------------------------------------------------------------------------

typedef __attribute__((ext_vector_type(16))) _Float16 v16h;
typedef __attribute__((ext_vector_type(8)))  float    v8f;

#define D_IN  128
#define D_OUT 64

// deg[i] = 1.0f  (self-loop contribution)
__global__ __launch_bounds__(256) void sgc_deg_init(float* __restrict__ deg, int N) {
    int i = blockIdx.x * 256 + threadIdx.x;
    if (i < N) deg[i] = 1.0f;
}

// deg[dst[e]] += 1 for every real edge
__global__ __launch_bounds__(256) void sgc_deg_count(const long long* __restrict__ dst,
                                                     float* __restrict__ deg, int E) {
    int e = blockIdx.x * 256 + threadIdx.x;
    if (e < E) atomicAdd(&deg[dst[e]], 1.0f);
}

// dinv[i] = rsqrt(deg[i])   (deg >= 1 always, so no zero guard needed)
__global__ __launch_bounds__(256) void sgc_dinv(float* __restrict__ deg, int N) {
    int i = blockIdx.x * 256 + threadIdx.x;
    if (i < N) deg[i] = rsqrtf(deg[i]);
}

// Self-loop term: hout[i,f] = dinv[i]^2 * hin[i,f]   (initializes hout)
__global__ __launch_bounds__(256) void sgc_self_scale(const float* __restrict__ dinv,
                                                      const float* __restrict__ hin,
                                                      float* __restrict__ hout, int total) {
    int idx = blockIdx.x * 256 + threadIdx.x;
    if (idx < total) {
        int node = idx >> 7;               // / D_IN
        float d = dinv[node];
        hout[idx] = d * d * hin[idx];
    }
}

// Edge scatter: hout[dst] += dinv[src]*dinv[dst] * hin[src]
// One wave (32 lanes) per edge; each lane moves a float4 (coalesced 512B row).
__global__ __launch_bounds__(256) void sgc_edge_prop(const long long* __restrict__ src,
                                                     const long long* __restrict__ dst,
                                                     const float* __restrict__ dinv,
                                                     const float* __restrict__ hin,
                                                     float* __restrict__ hout, int E) {
    long long g = (long long)blockIdx.x * 256 + threadIdx.x;
    int e = (int)(g >> 5);                 // wave-uniform edge id
    int lane = (int)(g & 31);
    if (e >= E) return;
    long long s = src[e];
    long long d = dst[e];
    float w = dinv[s] * dinv[d];
    const float4* hs = (const float4*)(hin + (size_t)s * D_IN);
    float4 v = hs[lane];                   // features 4*lane .. 4*lane+3
    float* ho = hout + (size_t)d * D_IN + lane * 4;
    atomicAdd(ho + 0, w * v.x);
    atomicAdd(ho + 1, w * v.y);
    atomicAdd(ho + 2, w * v.z);
    atomicAdd(ho + 3, w * v.w);
}

// emb[N,64] = h[N,128] @ W^T + b  via v_wmma_f32_16x16x32_f16.
// 8 waves/block: wave w -> row-tile (w>>2) of 2, col-tile (w&3) of 4.
// Fragment layouts used here (contiguous k order, same for A and B):
//   A (16x32 f16): lane m=lane&15; K run  [ka,ka+16), ka=kk+16*laneHi
//   B (32x16 f16): lane n=lane&15; K run  [kb,kb+16), kb=kk+16*laneHi
//   C/D (16x16 f32): c[v] -> row rowBase+v+8*laneHi, col colBase+(lane&15)
__global__ __launch_bounds__(256) void sgc_gemm_wmma(const float* __restrict__ h,
                                                     const float* __restrict__ W,
                                                     const float* __restrict__ bias,
                                                     float* __restrict__ emb, int N) {
    int lane   = threadIdx.x & 31;
    int wave   = threadIdx.x >> 5;
    int laneHi = lane >> 4;
    int laneLo = lane & 15;
    int rowBase = blockIdx.x * 32 + (wave >> 2) * 16;
    if (rowBase >= N) return;              // N is a multiple of 32 here: full tiles
    int colBase = (wave & 3) * 16;

    const float* hrow = h + (size_t)(rowBase + laneLo) * D_IN;  // A row for this lane
    const float* wrow = W + (size_t)(colBase + laneLo) * D_IN;  // B col n = W row n

    v8f c = {};
#pragma unroll
    for (int kk = 0; kk < D_IN; kk += 32) {
        v16h av, bv;
        int ka = kk + 16 * laneHi;
#pragma unroll
        for (int j = 0; j < 16; ++j)
            av[j] = (_Float16)hrow[ka + j];
        int kb = kk + 16 * laneHi;
#pragma unroll
        for (int j = 0; j < 16; ++j)
            bv[j] = (_Float16)wrow[kb + j];
        // (neg_a, A, neg_b, B, c_mod, C, reuse_a, reuse_b)
        c = __builtin_amdgcn_wmma_f32_16x16x32_f16(false, av, false, bv,
                                                   (short)0, c, false, false);
    }

    int n = colBase + laneLo;
    float bn = bias[n];
#pragma unroll
    for (int v = 0; v < 8; ++v) {
        int m = rowBase + v + 8 * laneHi;
        emb[(size_t)m * D_OUT + n] = c[v] + bn;
    }
}

// out[r,:] = emb[r,:] - max - log(sum(exp(emb - max))), 64 cols = 2 per lane.
__global__ __launch_bounds__(256) void sgc_log_softmax(const float* __restrict__ emb,
                                                       float* __restrict__ out, int N) {
    int wave = threadIdx.x >> 5;
    int lane = threadIdx.x & 31;
    int r = blockIdx.x * 8 + wave;
    if (r >= N) return;
    const float* er = emb + (size_t)r * D_OUT;
    float e0 = er[lane];
    float e1 = er[lane + 32];
    float m = fmaxf(e0, e1);
#pragma unroll
    for (int off = 16; off > 0; off >>= 1)
        m = fmaxf(m, __shfl_xor(m, off));
    float s = __expf(e0 - m) + __expf(e1 - m);
#pragma unroll
    for (int off = 16; off > 0; off >>= 1)
        s += __shfl_xor(s, off);
    float lse = __logf(s);
    float* orow = out + (size_t)r * D_OUT;
    orow[lane]      = e0 - m - lse;
    orow[lane + 32] = e1 - m - lse;
}

extern "C" void kernel_launch(void* const* d_in, const int* in_sizes, int n_in,
                              void* d_out, int out_size, void* d_ws, size_t ws_size,
                              hipStream_t stream) {
    const float*     x    = (const float*)d_in[0];       // [N,128]
    const long long* eidx = (const long long*)d_in[1];   // [2,E] int64
    const float*     W    = (const float*)d_in[2];       // [64,128]
    const float*     bias = (const float*)d_in[3];       // [64]

    const int N = in_sizes[0] / D_IN;
    const int E = in_sizes[1] / 2;
    const long long* src = eidx;
    const long long* dst = eidx + E;

    // Workspace: dinv[N] | h1[N*128] | h2[N*128]   (~103 MB)
    float* dinv = (float*)d_ws;
    float* h1   = dinv + N;
    float* h2   = h1 + (size_t)N * D_IN;

    float* out_ls = (float*)d_out;                       // log_softmax output
    float* emb    = out_ls + (size_t)N * D_OUT;          // raw embeddings output

    const int total = N * D_IN;

    // Degree + normalization
    sgc_deg_init <<<(N + 255) / 256, 256, 0, stream>>>(dinv, N);
    sgc_deg_count<<<(E + 255) / 256, 256, 0, stream>>>(dst, dinv, E);
    sgc_dinv     <<<(N + 255) / 256, 256, 0, stream>>>(dinv, N);

    // Hop 1: x -> h1
    sgc_self_scale<<<(total + 255) / 256, 256, 0, stream>>>(dinv, x, h1, total);
    sgc_edge_prop <<<(int)(((long long)E * 32 + 255) / 256), 256, 0, stream>>>(
        src, dst, dinv, x, h1, E);

    // Hop 2: h1 -> h2
    sgc_self_scale<<<(total + 255) / 256, 256, 0, stream>>>(dinv, h1, h2, total);
    sgc_edge_prop <<<(int)(((long long)E * 32 + 255) / 256), 256, 0, stream>>>(
        src, dst, dinv, h1, h2, E);

    // Projection (WMMA) + bias
    sgc_gemm_wmma<<<(N + 31) / 32, 256, 0, stream>>>(h2, W, bias, emb, N);

    // log_softmax
    sgc_log_softmax<<<(N + 7) / 8, 256, 0, stream>>>(emb, out_ls, N);
}
